// MultiHeadSelfAttention_38104949850174
// MI455X (gfx1250) — matrix operand order repaired and checked by reading, not compiled, not run on hardware
//
#include <hip/hip_runtime.h>

#define S_LEN   4096
#define D_MODEL 768
#define N_HEADS 12
#define D_HEAD  64

typedef __attribute__((ext_vector_type(16))) __bf16       v16bf;
typedef __attribute__((ext_vector_type(8)))  float        v8f;
typedef __attribute__((ext_vector_type(4)))  unsigned int v4u;
typedef __attribute__((ext_vector_type(2)))  unsigned int v2u;
typedef __attribute__((ext_vector_type(4)))  float        v4f;

// ---------- helpers ----------

__device__ __forceinline__ unsigned short f2bf(float f) {
    unsigned int u = __float_as_uint(f);
    u += 0x7FFFu + ((u >> 16) & 1u);          // round-to-nearest-even
    return (unsigned short)(u >> 16);
}

// A-matrix 16x32 bf16 fragment (ISA 7.12.2): lane holds row M=lane&15,
// half 0 -> K {base..+7, base+16..+23}, half 1 -> +8.
__device__ __forceinline__ v16bf load_frag_a(const unsigned short* p) {
    union { v16bf v; v4u q[2]; } f;
    f.q[0] = *(const v4u*)(p);
    f.q[1] = *(const v4u*)(p + 16);
    return f.v;
}

// B-matrix 32x16 bf16 fragment: lane holds column N=lane&15,
// half 0 -> K {base..+7, base+16..+23}, half 1 -> +8.
__device__ __forceinline__ v16bf load_frag_b(const unsigned short* p) {
    union { v16bf v; v4u q[2]; } f;
    f.q[0] = *(const v4u*)(p);
    f.q[1] = *(const v4u*)(p + 16);
    return f.v;
}

__device__ __forceinline__ v8f bwmma(v16bf a, v16bf b, v8f c) {
    return __builtin_amdgcn_wmma_f32_16x16x32_bf16(
        false, a, false, b, (short)0, c, false, false);
}

// ---- CDNA5 async global->LDS (ASYNCcnt path, bypasses VGPRs) ----

__device__ __forceinline__ unsigned lds_off(const void* p) {
    // LDS aperture: flat address truncated to [31:0] is the LDS byte address
    return (unsigned)(unsigned long long)p;
}

__device__ __forceinline__ void async_g2l_16B(unsigned lds_addr, const void* g) {
    asm volatile("global_load_async_to_lds_b128 %0, %1, off"
                 :
                 : "v"(lds_addr), "v"(g)
                 : "memory");
}

__device__ __forceinline__ void wait_async0() {
    asm volatile("s_wait_asynccnt 0x0" ::: "memory");
}

// ---------- prep kernels ----------

__global__ void cvt_x_kernel(const float* __restrict__ x,
                             unsigned short* __restrict__ xh) {
    int i = (blockIdx.x * blockDim.x + threadIdx.x) * 4;
    v4f f = *(const v4f*)(x + i);
    unsigned int lo = (unsigned int)f2bf(f.x) | ((unsigned int)f2bf(f.y) << 16);
    unsigned int hi = (unsigned int)f2bf(f.z) | ((unsigned int)f2bf(f.w) << 16);
    *(v2u*)(xh + i) = (v2u){lo, hi};
}

// W stored [K][N] (in,out); produce WT [N][K] bf16 so B-fragments are contiguous.
__global__ void prep_w_kernel(const float* __restrict__ W,
                              unsigned short* __restrict__ WT) {
    int idx  = blockIdx.x * blockDim.x + threadIdx.x;  // idx = n*K + k
    int nrow = idx / D_MODEL;
    int kcol = idx - nrow * D_MODEL;
    WT[idx] = f2bf(W[kcol * D_MODEL + nrow]);
}

// ---------- WMMA GEMM:  C = A[M,K] * BT[N,K]^T + bias ----------
// 256 threads = 8 waves splitting M (128 rows); all share one 64-wide B tile
// that is double-buffered in LDS via async global->LDS loads.
// mode 0: Q -> bf16 [H][S][64] * 1/8 | 1: K -> [H][S][64] | 2: V -> [H][64][S]
// mode 3: fp32 out [M][768]
__global__ __launch_bounds__(256)
void gemm_bf16_kernel(const unsigned short* __restrict__ A,
                      const unsigned short* __restrict__ BT,
                      const float* __restrict__ bias,
                      float* __restrict__ outF,
                      unsigned short* __restrict__ outH,
                      int mode) {
    __shared__ unsigned short Bt[2][64 * 64];   // 2 x 8KB double buffer

    const int K = D_MODEL;
    const int tid  = threadIdx.x;
    const int lane = tid & 31;
    const int wid  = tid >> 5;
    const int m0 = blockIdx.x * 128 + wid * 16;
    const int n0 = blockIdx.y * 64;
    const int r  = lane & 15;
    const int hi = lane >> 4;

    const unsigned short* ap = A + (size_t)(m0 + r) * K + hi * 8;

    // stage one 64(N) x 64(K) tile of BT: 64 rows x 128B, row stride K*2 bytes
    auto stage_b = [&](int buf, int k0) {
        unsigned base = lds_off(&Bt[buf][0]);
#pragma unroll
        for (int s2 = 0; s2 < 2; ++s2) {
            int seg = tid + s2 * 256;           // 0..511
            int row = seg >> 3;
            int c   = (seg & 7) * 16;           // byte offset in row
            async_g2l_16B(base + row * 128 + c,
                          (const char*)(BT + (size_t)(n0 + row) * K + k0) + c);
        }
    };

    stage_b(0, 0);

    v8f acc[4] = {};
    for (int c = 0; c < K / 64; ++c) {
        const int buf = c & 1;
        wait_async0();
        __syncthreads();
        if (c + 1 < K / 64) stage_b(buf ^ 1, (c + 1) * 64);

        const unsigned short* bl = &Bt[buf][0];
#pragma unroll
        for (int ks = 0; ks < 2; ++ks) {
            v16bf a = load_frag_a(ap + c * 64 + ks * 32);
#pragma unroll
            for (int nt = 0; nt < 4; ++nt) {
                v16bf b = load_frag_b(bl + (nt * 16 + r) * 64 + ks * 32 + hi * 8);
                acc[nt] = bwmma(a, b, acc[nt]);
            }
        }
    }

    const int mB = m0 + 8 * hi;   // C layout: vgpr i -> row mB+i, col n0+nt*16+r
#pragma unroll
    for (int nt = 0; nt < 4; ++nt) {
        int n = n0 + nt * 16 + r;
        float bval = bias[n];
        int h = n >> 6, d = n & 63;
#pragma unroll
        for (int i = 0; i < 8; ++i) {
            int m = mB + i;
            float v = acc[nt][i] + bval;
            if (mode == 3) {
                outF[(size_t)m * D_MODEL + n] = v;
            } else if (mode == 0) {
                outH[((size_t)h * S_LEN + m) * D_HEAD + d] = f2bf(v * 0.125f);
            } else if (mode == 1) {
                outH[((size_t)h * S_LEN + m) * D_HEAD + d] = f2bf(v);
            } else {
                outH[((size_t)h * D_HEAD + d) * S_LEN + m] = f2bf(v);
            }
        }
    }
}

// ---------- flash attention ----------
// 256 threads = 8 waves, each owning one 16-row query block of the same head.
// All 8 waves share K/V 64-key tiles double-buffered in LDS (async staged).
__global__ __launch_bounds__(256)
void flash_attn_kernel(const unsigned short* __restrict__ Q,   // [H][S][64] prescaled
                       const unsigned short* __restrict__ Km,  // [H][S][64]
                       const unsigned short* __restrict__ VT,  // [H][64][S]
                       unsigned short* __restrict__ Aout) {    // bf16 [S][768]
    __shared__ unsigned short Kt[2][64 * 64];   // [local key][d]   2 x 8KB
    __shared__ unsigned short Vt[2][64 * 64];   // [d][local key]   2 x 8KB
    __shared__ unsigned short Pl[8][16 * 64];   // per-wave P stage  16KB

    const int tid  = threadIdx.x;
    const int lane = tid & 31;
    const int wid  = tid >> 5;
    const int h    = blockIdx.y;
    const int q0   = (blockIdx.x * 8 + wid) * 16;
    const int r    = lane & 15;
    const int hi   = lane >> 4;

    const unsigned short* qp = Q + ((size_t)h * S_LEN + q0 + r) * D_HEAD + hi * 8;
    v16bf qa0 = load_frag_a(qp);
    v16bf qa1 = load_frag_a(qp + 32);

    const unsigned short* kg = Km + (size_t)h * S_LEN * D_HEAD;   // rows contiguous
    const unsigned short* vg = VT + (size_t)h * D_HEAD * S_LEN;   // 64 rows, stride S
    unsigned short* pl = &Pl[wid][0];

    // stage 64-key K tile (contiguous 8KB) and V tile (64 rows x 128B)
    auto stage_kv = [&](int buf, int kk) {
        unsigned kb = lds_off(&Kt[buf][0]);
        unsigned vb = lds_off(&Vt[buf][0]);
        const char* ks = (const char*)(kg + (size_t)kk * D_HEAD);
#pragma unroll
        for (int s2 = 0; s2 < 2; ++s2) {
            int off = (tid + s2 * 256) * 16;
            async_g2l_16B(kb + off, ks + off);
        }
#pragma unroll
        for (int s2 = 0; s2 < 2; ++s2) {
            int seg = tid + s2 * 256;           // 0..511
            int row = seg >> 3;
            int c   = (seg & 7) * 16;
            async_g2l_16B(vb + row * 128 + c,
                          (const char*)(vg + (size_t)row * S_LEN + kk) + c);
        }
    };

    stage_kv(0, 0);

    v8f ov[4] = {};
    float mrow[8], lrow[8];
#pragma unroll
    for (int i = 0; i < 8; ++i) { mrow[i] = -3.0e38f; lrow[i] = 0.0f; }

    for (int kk = 0; kk < S_LEN; kk += 64) {
        const int buf = (kk >> 6) & 1;
        wait_async0();
        __syncthreads();
        if (kk + 64 < S_LEN) stage_kv(buf ^ 1, kk + 64);

        const unsigned short* kl = &Kt[buf][0];
        const unsigned short* vl = &Vt[buf][0];

        // ---- S = Q * K^T (4 tiles of 16 keys) from LDS ----
        v8f st[4];
#pragma unroll
        for (int nt = 0; nt < 4; ++nt) {
            const unsigned short* bp = kl + (nt * 16 + r) * 64 + hi * 8;
            v8f c = {};
            c = bwmma(qa0, load_frag_b(bp), c);
            c = bwmma(qa1, load_frag_b(bp + 32), c);
            st[nt] = c;
        }

        // ---- online softmax: row max via xor-shuffles (per lane-half rows) ----
        float mnew[8];
#pragma unroll
        for (int i = 0; i < 8; ++i) mnew[i] = mrow[i];
#pragma unroll
        for (int nt = 0; nt < 4; ++nt)
#pragma unroll
            for (int i = 0; i < 8; ++i) mnew[i] = fmaxf(mnew[i], st[nt][i]);
#pragma unroll
        for (int i = 0; i < 8; ++i) {
            mnew[i] = fmaxf(mnew[i], __shfl_xor(mnew[i], 1));
            mnew[i] = fmaxf(mnew[i], __shfl_xor(mnew[i], 2));
            mnew[i] = fmaxf(mnew[i], __shfl_xor(mnew[i], 4));
            mnew[i] = fmaxf(mnew[i], __shfl_xor(mnew[i], 8));
        }

        float psum[8];
#pragma unroll
        for (int i = 0; i < 8; ++i) {
            float sc = __expf(mrow[i] - mnew[i]);
            lrow[i] *= sc;
            psum[i] = 0.0f;
            mrow[i] = mnew[i];
#pragma unroll
            for (int nt = 0; nt < 4; ++nt) ov[nt][i] *= sc;
        }

        // ---- P = exp(S - m), stage to LDS (C layout -> row-major 16x64) ----
#pragma unroll
        for (int nt = 0; nt < 4; ++nt)
#pragma unroll
            for (int i = 0; i < 8; ++i) {
                float p = __expf(st[nt][i] - mnew[i]);
                psum[i] += p;
                pl[(i + 8 * hi) * 64 + nt * 16 + r] = f2bf(p);
            }
#pragma unroll
        for (int i = 0; i < 8; ++i) {
            float s = psum[i];
            s += __shfl_xor(s, 1);
            s += __shfl_xor(s, 2);
            s += __shfl_xor(s, 4);
            s += __shfl_xor(s, 8);
            lrow[i] += s;
        }

        // ---- O += P * V : P as A-fragments from LDS, V tile from LDS ----
        const unsigned short* pp = pl + r * 64 + hi * 8;
        v16bf pa0 = load_frag_a(pp);
        v16bf pa1 = load_frag_a(pp + 32);
#pragma unroll
        for (int nt = 0; nt < 4; ++nt) {
            const unsigned short* vp = vl + (nt * 16 + r) * 64 + hi * 8;
            ov[nt] = bwmma(pa0, load_frag_b(vp), ov[nt]);
            ov[nt] = bwmma(pa1, load_frag_b(vp + 32), ov[nt]);
        }
    }

    // ---- epilogue: O / l, write bf16 [S][768] ----
    float linv[8];
#pragma unroll
    for (int i = 0; i < 8; ++i) linv[i] = 1.0f / lrow[i];
    const int mB = q0 + 8 * hi;
#pragma unroll
    for (int nt = 0; nt < 4; ++nt) {
        int n = h * 64 + nt * 16 + r;
#pragma unroll
        for (int i = 0; i < 8; ++i)
            Aout[(size_t)(mB + i) * D_MODEL + n] = f2bf(ov[nt][i] * linv[i]);
    }
}

// ---------- launch ----------

extern "C" void kernel_launch(void* const* d_in, const int* in_sizes, int n_in,
                              void* d_out, int out_size, void* d_ws, size_t ws_size,
                              hipStream_t stream) {
    const float* x  = (const float*)d_in[0];
    const float* Wq = (const float*)d_in[1];
    const float* bq = (const float*)d_in[2];
    const float* Wk = (const float*)d_in[3];
    const float* bk = (const float*)d_in[4];
    const float* Wv = (const float*)d_in[5];
    const float* bv = (const float*)d_in[6];
    const float* Wo = (const float*)d_in[7];
    const float* bo = (const float*)d_in[8];
    float* out = (float*)d_out;

    const size_t SZ_X = (size_t)S_LEN * D_MODEL;      // 3,145,728 elems
    const size_t SZ_W = (size_t)D_MODEL * D_MODEL;    //   589,824 elems
    unsigned short* xh  = (unsigned short*)d_ws;
    unsigned short* WqT = xh  + SZ_X;
    unsigned short* WkT = WqT + SZ_W;
    unsigned short* WvT = WkT + SZ_W;
    unsigned short* WoT = WvT + SZ_W;
    unsigned short* Qh  = WoT + SZ_W;                 // [H][S][64]
    unsigned short* Kh  = Qh  + SZ_X;                 // [H][S][64]
    unsigned short* VTh = Kh  + SZ_X;                 // [H][64][S]
    unsigned short* Ah  = VTh + SZ_X;                 // bf16 [S][768]

    // 1) precision conversion / weight transpose
    cvt_x_kernel<<<(SZ_X / 4 + 255) / 256, 256, 0, stream>>>(x, xh);
    prep_w_kernel<<<SZ_W / 256, 256, 0, stream>>>(Wq, WqT);
    prep_w_kernel<<<SZ_W / 256, 256, 0, stream>>>(Wk, WkT);
    prep_w_kernel<<<SZ_W / 256, 256, 0, stream>>>(Wv, WvT);
    prep_w_kernel<<<SZ_W / 256, 256, 0, stream>>>(Wo, WoT);

    // 2) QKV projections (WMMA, async-staged B tiles)
    dim3 ggrid(S_LEN / 128, D_MODEL / 64);
    gemm_bf16_kernel<<<ggrid, 256, 0, stream>>>(xh, WqT, bq, nullptr, Qh, 0);
    gemm_bf16_kernel<<<ggrid, 256, 0, stream>>>(xh, WkT, bk, nullptr, Kh, 1);
    gemm_bf16_kernel<<<ggrid, 256, 0, stream>>>(xh, WvT, bv, nullptr, VTh, 2);

    // 3) flash attention (WMMA QK^T + PV, async-staged K/V tiles)
    flash_attn_kernel<<<dim3(S_LEN / 16 / 8, N_HEADS), 256, 0, stream>>>(Qh, Kh, VTh, Ah);

    // 4) output projection -> fp32 + bias
    gemm_bf16_kernel<<<ggrid, 256, 0, stream>>>(Ah, WoT, bo, out, nullptr, 3);
}
